// BiRNNCrf_11879879542083
// MI455X (gfx1250) — matrix operand order repaired and checked by reading, not compiled, not run on hardware
//
#include <hip/hip_runtime.h>
#include <hip/hip_bf16.h>

// ---------------------------------------------------------------------------
// BiLSTM + FC for MI455X (gfx1250, wave32, WMMA f32_16x16x32_f16).
// Big GEMMs + recurrent h@U through v_wmma; A tiles staged in LDS via
// CDNA5 async global->LDS copies (ASYNCcnt); weights pre-packed into
// B-fragment layout; LSTM state kept f32.
// ---------------------------------------------------------------------------

#define BATCH 64
#define TLEN  512
#define EMB   300
#define HID   300
#define G4H   1200           // 4*HID
#define TAGS  64
#define ROWS  (BATCH*TLEN)   // 32768
#define KPAD  320            // EMB/HID padded to mult of 32
#define KT_K  10             // KPAD/32
#define NT_W  75             // 1200/16
#define NSTRIP 19            // ceil(300/16)
#define KCAT  608            // 2*HID padded to mult of 32
#define KT_C  19             // KCAT/32

typedef __attribute__((ext_vector_type(16))) _Float16 v16h;
typedef __attribute__((ext_vector_type(8)))  _Float16 v8h;
typedef __attribute__((ext_vector_type(8)))  float    v8f;
typedef __attribute__((ext_vector_type(4)))  float    f4;
typedef int v4i __attribute__((vector_size(4 * sizeof(int))));

#define AS1 __attribute__((address_space(1)))
#define AS3 __attribute__((address_space(3)))

#if __has_builtin(__builtin_amdgcn_global_load_async_to_lds_b128) && \
    __has_builtin(__builtin_amdgcn_s_wait_asynccnt)
#define HAVE_ASYNC_LDS 1
#else
#define HAVE_ASYNC_LDS 0
#endif

// 16B global -> LDS, async (ASYNCcnt) when the toolchain exposes it.
__device__ __forceinline__ void copy16_to_lds(const _Float16* g, _Float16* l) {
#if HAVE_ASYNC_LDS
  __builtin_amdgcn_global_load_async_to_lds_b128(
      (AS1 v4i*)(uintptr_t)g,
      (AS3 v4i*)(uint32_t)(uintptr_t)l,
      0, 0);
#else
  *(v8h*)l = *(const v8h*)g;
#endif
}

__device__ __forceinline__ void lds_copy_join() {
#if HAVE_ASYNC_LDS
  __builtin_amdgcn_s_wait_asynccnt(0);
#endif
  __syncthreads();
}

// ---- fragment loaders ------------------------------------------------------
// 16-bit A 16x32 striping: lane l holds row (l&15); halves 0..15 = K kb..kb+15,
// kb = (l>>4)*16.  A tile lives in LDS,
// row-major [16][KPAD].
__device__ __forceinline__ v16h load_a16_lds(const _Float16* A, int k0) {
  const int l = threadIdx.x & 31;
  const _Float16* p = A + (l & 15) * KPAD + k0 + ((l >> 4) << 4);
  v8h lo = *(const v8h*)(p);
  v8h hi = *(const v8h*)(p + 8);
  v16h a;
#pragma unroll
  for (int i = 0; i < 8; ++i) { a[i] = lo[i]; a[i + 8] = hi[i]; }
  return a;
}

// A from global (used by gemm_fc), leading dim lda.
__device__ __forceinline__ v16h load_a16_g(const _Float16* __restrict__ A,
                                           int lda, int m0, int k0) {
  const int l = threadIdx.x & 31;
  const _Float16* p = A + (size_t)(m0 + (l & 15)) * lda + k0 + ((l >> 4) << 4);
  v8h lo = *(const v8h*)(p);
  v8h hi = *(const v8h*)(p + 8);
  v16h a;
#pragma unroll
  for (int i = 0; i < 8; ++i) { a[i] = lo[i]; a[i + 8] = hi[i]; }
  return a;
}

// B fragments pre-packed: per tile, lane l owns 16 contiguous halves
// (k_local = (l>>4)*16 + h, n_local = l&15).
__device__ __forceinline__ v16h load_bpk(const _Float16* __restrict__ P) {
  const int l = threadIdx.x & 31;
  const _Float16* p = P + l * 16;
  v8h lo = *(const v8h*)(p);
  v8h hi = *(const v8h*)(p + 8);
  v16h b;
#pragma unroll
  for (int i = 0; i < 8; ++i) { b[i] = lo[i]; b[i + 8] = hi[i]; }
  return b;
}

__device__ __forceinline__ v8f wmma_acc(v16h a, v16h b, v8f c) {
  return __builtin_amdgcn_wmma_f32_16x16x32_f16(false, a, false, b,
                                                (short)0, c, false, false);
}

__device__ __forceinline__ float sigm(float x) {
  return 1.0f / (1.0f + __expf(-x));
}

// ---- init / pack / gather --------------------------------------------------
__global__ void zero_ws(f4* __restrict__ p, long n4) {
  for (long i = blockIdx.x * (long)blockDim.x + threadIdx.x; i < n4;
       i += (long)gridDim.x * blockDim.x) {
    f4 z = {0.f, 0.f, 0.f, 0.f};
    p[i] = z;
  }
}

// Wpk layout: [dir][kt(10)][nt(75)][512]
__global__ void pack_W(const float* __restrict__ Wf, const float* __restrict__ Wb,
                       _Float16* __restrict__ Wpk) {
  const long total = 2L * KT_K * NT_W * 512;
  for (long idx = blockIdx.x * (long)blockDim.x + threadIdx.x; idx < total;
       idx += (long)gridDim.x * blockDim.x) {
    int dir = (int)(idx / (KT_K * NT_W * 512));
    int r   = (int)(idx % (KT_K * NT_W * 512));
    int kt = r / (NT_W * 512); int r2 = r % (NT_W * 512);
    int nt = r2 / 512;         int e  = r2 % 512;
    int lane = e >> 4, h = e & 15;
    int k = kt * 32 + ((lane >> 4) << 4) + h;
    int n = nt * 16 + (lane & 15);
    const float* W = dir ? Wb : Wf;
    Wpk[idx] = (_Float16)((k < EMB) ? W[(size_t)k * G4H + n] : 0.f);
  }
}

// Upk layout: [dir][gate(4)][kt(10)][strip(19)][512], gate-local columns
__global__ void pack_U(const float* __restrict__ Uf, const float* __restrict__ Ub,
                       _Float16* __restrict__ Upk) {
  const long total = 2L * 4 * KT_K * NSTRIP * 512;
  for (long idx = blockIdx.x * (long)blockDim.x + threadIdx.x; idx < total;
       idx += (long)gridDim.x * blockDim.x) {
    int dir = (int)(idx / (4L * KT_K * NSTRIP * 512));
    long r  = idx % (4L * KT_K * NSTRIP * 512);
    int gate = (int)(r / (KT_K * NSTRIP * 512));
    int r2 = (int)(r % (KT_K * NSTRIP * 512));
    int kt = r2 / (NSTRIP * 512); int r3 = r2 % (NSTRIP * 512);
    int strip = r3 / 512;         int e  = r3 % 512;
    int lane = e >> 4, h = e & 15;
    int k  = kt * 32 + ((lane >> 4) << 4) + h;
    int nl = strip * 16 + (lane & 15);
    const float* U = dir ? Ub : Uf;
    float v = (k < HID && nl < HID) ? U[(size_t)k * G4H + gate * HID + nl] : 0.f;
    Upk[idx] = (_Float16)v;
  }
}

// Wfcpk layout: [kt(19)][nt(4)][512]
__global__ void pack_Wfc(const float* __restrict__ Wfc, _Float16* __restrict__ Wpk) {
  const long total = (long)KT_C * 4 * 512;
  for (long idx = blockIdx.x * (long)blockDim.x + threadIdx.x; idx < total;
       idx += (long)gridDim.x * blockDim.x) {
    int kt = (int)(idx / (4 * 512));
    int r  = (int)(idx % (4 * 512));
    int nt = r / 512; int e = r % 512;
    int lane = e >> 4, h = e & 15;
    int k = kt * 32 + ((lane >> 4) << 4) + h;
    int n = nt * 16 + (lane & 15);
    Wpk[idx] = (_Float16)((k < 2 * HID) ? Wfc[(size_t)k * TAGS + n] : 0.f);
  }
}

// X16[row][320] = emb[tokens[row]][k] as f16, K zero-padded
__global__ void gather_x(const int* __restrict__ tokens,
                         const float* __restrict__ emb,
                         _Float16* __restrict__ X16) {
  const long total = (long)ROWS * KPAD;
  for (long idx = blockIdx.x * (long)blockDim.x + threadIdx.x; idx < total;
       idx += (long)gridDim.x * blockDim.x) {
    int row = (int)(idx / KPAD);
    int k   = (int)(idx % KPAD);
    float v = 0.f;
    if (k < EMB) v = emb[(size_t)tokens[row] * EMB + k];
    X16[idx] = (_Float16)v;
  }
}

// ---- xg = X @ W + b  (both directions), stores f16 [2][ROWS][1200] ---------
// Block: 8 waves share one 16x320 A tile staged in LDS (async copy);
// each wave owns one N tile.
__global__ void gemm_xg(const _Float16* __restrict__ X16,
                        const _Float16* __restrict__ Wpk,
                        const float* __restrict__ bf,
                        const float* __restrict__ bb,
                        _Float16* __restrict__ xg) {
  __shared__ _Float16 At[16 * KPAD];          // 10 KB
  const int mt  = blockIdx.y;                 // 0..2047
  const int dir = blockIdx.z;
  const int nt  = blockIdx.x * 8 + (threadIdx.x >> 5);
  const int l   = threadIdx.x & 31;
  const int m0  = mt * 16;

  // cooperative A-tile stage: 640 x 16B chunks, 256 threads
  const _Float16* Ab = X16 + (size_t)m0 * KPAD;
  for (int c = threadIdx.x; c < 16 * (KPAD / 8); c += 256) {
    int row = c / (KPAD / 8);
    int off = (c % (KPAD / 8)) * 8;
    copy16_to_lds(Ab + (size_t)row * KPAD + off, &At[row * KPAD + off]);
  }
  lds_copy_join();

  if (nt < NT_W) {
    const _Float16* Wd = Wpk + (size_t)dir * (KT_K * NT_W * 512);
    v8f acc = {};
#pragma unroll
    for (int kt = 0; kt < KT_K; ++kt) {
      if (kt + 1 < KT_K)
        __builtin_prefetch(Wd + (size_t)((kt + 1) * NT_W + nt) * 512, 0, 0);
      v16h a = load_a16_lds(At, kt * 32);
      v16h b = load_bpk(Wd + (size_t)(kt * NT_W + nt) * 512);
      acc = wmma_acc(a, b, acc);
    }
    const int n = nt * 16 + (l & 15);
    const float bv = (dir ? bb : bf)[n];
    _Float16* o = xg + ((size_t)dir * ROWS + m0) * G4H + n;
    const int rbase = (l >> 4) * 8;
#pragma unroll
    for (int v = 0; v < 8; ++v)
      o[(size_t)(rbase + v) * G4H] = (_Float16)(acc[v] + bv);
  }
}

// ---- one recurrent step (both directions), 4 waves = 4 gate tiles ----------
__global__ void lstm_step(const _Float16* __restrict__ xg,
                          const _Float16* __restrict__ Upk,
                          const float* __restrict__ bf,
                          const float* __restrict__ bb,
                          const int* __restrict__ seq_len,
                          float* __restrict__ h_state,   // [2][64][300]
                          float* __restrict__ c_state,   // [2][64][300]
                          _Float16* __restrict__ h16,    // [2][64][320]
                          _Float16* __restrict__ hcat,   // [ROWS][608]
                          int s) {
  __shared__ _Float16 Ht[16 * KPAD];       // 10 KB: shared h tile
  __shared__ float g[4][16][16];           // 4 KB: gate pre-activations
  const int strip = blockIdx.x;            // 0..18
  const int mt    = blockIdx.y;            // 0..3
  const int dir   = blockIdx.z;
  const int gate  = threadIdx.x >> 5;      // 0..3 (wave id)
  const int l     = threadIdx.x & 31;
  const int t     = dir ? (TLEN - 1 - s) : s;
  const int m0    = mt * 16;

  // stage 16x320 h tile once per block (all 4 gate waves consume it)
  const _Float16* hA = h16 + ((size_t)dir * BATCH + m0) * KPAD;
  for (int c = threadIdx.x; c < 16 * (KPAD / 8); c += 128) {
    int row = c / (KPAD / 8);
    int off = (c % (KPAD / 8)) * 8;
    copy16_to_lds(hA + (size_t)row * KPAD + off, &Ht[row * KPAD + off]);
  }
  lds_copy_join();

  const _Float16* Ug = Upk + (size_t)(dir * 4 + gate) * (KT_K * NSTRIP * 512);
  v8f acc = {};
#pragma unroll
  for (int kt = 0; kt < KT_K; ++kt) {
    v16h a = load_a16_lds(Ht, kt * 32);
    v16h b = load_bpk(Ug + (size_t)(kt * NSTRIP + strip) * 512);
    acc = wmma_acc(a, b, acc);
  }

  const int col = l & 15;
  const int n   = strip * 16 + col;
  const float bv = (n < HID) ? (dir ? bb : bf)[gate * HID + n] : 0.f;
  const int rbase = (l >> 4) * 8;
#pragma unroll
  for (int v = 0; v < 8; ++v) {
    int r = rbase + v;
    float val = acc[v];
    if (n < HID) {
      int bi = m0 + r;
      val += bv + (float)xg[(((size_t)dir * ROWS) + (size_t)bi * TLEN + t) * G4H
                            + gate * HID + n];
    }
    g[gate][r][col] = val;
  }
  __syncthreads();

#pragma unroll
  for (int e = threadIdx.x; e < 256; e += 128) {
    int r = e >> 4, cc = e & 15;
    int n2 = strip * 16 + cc;
    if (n2 < HID) {
      int bi = m0 + r;
      float gi = g[0][r][cc], gf = g[1][r][cc];
      float gc = g[2][r][cc], go = g[3][r][cc];
      float ii = sigm(gi), ff = sigm(gf), oo = sigm(go);
      size_t si = ((size_t)dir * BATCH + bi) * HID + n2;
      float c_old = c_state[si], h_old = h_state[si];
      float c_new = ff * c_old + ii * tanhf(gc);
      float h_new = oo * tanhf(c_new);
      bool msk = t < seq_len[bi];
      float hs = msk ? h_new : h_old;
      float cs = msk ? c_new : c_old;
      c_state[si] = cs;
      h_state[si] = hs;
      h16[((size_t)dir * BATCH + bi) * KPAD + n2] = (_Float16)hs;
      hcat[((size_t)bi * TLEN + t) * KCAT + dir * HID + n2] = (_Float16)hs;
    }
  }
}

// ---- out = hcat @ Wfc + bfc -------------------------------------------------
__global__ void gemm_fc(const _Float16* __restrict__ hcat,
                        const _Float16* __restrict__ Wfcpk,
                        const float* __restrict__ bfc,
                        float* __restrict__ out) {
  const int nt = blockIdx.x;                        // 0..3
  const int mt = blockIdx.y * 8 + (threadIdx.x >> 5);
  const int l  = threadIdx.x & 31;
  const int m0 = mt * 16;
  v8f acc = {};
#pragma unroll
  for (int kt = 0; kt < KT_C; ++kt) {
    v16h a = load_a16_g(hcat, KCAT, m0, kt * 32);
    v16h b = load_bpk(Wfcpk + (size_t)(kt * 4 + nt) * 512);
    acc = wmma_acc(a, b, acc);
  }
  const int n = nt * 16 + (l & 15);
  const float bv = bfc[n];
  const int rbase = (l >> 4) * 8;
#pragma unroll
  for (int v = 0; v < 8; ++v)
    out[(size_t)(m0 + rbase + v) * TAGS + n] = acc[v] + bv;
}

// ---------------------------------------------------------------------------
extern "C" void kernel_launch(void* const* d_in, const int* in_sizes, int n_in,
                              void* d_out, int out_size, void* d_ws, size_t ws_size,
                              hipStream_t stream) {
  (void)in_sizes; (void)n_in; (void)out_size; (void)ws_size;
  const int*   tokens  = (const int*)d_in[0];
  const int*   seq_len = (const int*)d_in[1];
  const float* emb     = (const float*)d_in[2];
  const float* Wf      = (const float*)d_in[3];
  const float* Uf      = (const float*)d_in[4];
  const float* bf      = (const float*)d_in[5];
  const float* Wb      = (const float*)d_in[6];
  const float* Ub      = (const float*)d_in[7];
  const float* bb      = (const float*)d_in[8];
  const float* Wfc     = (const float*)d_in[9];
  const float* bfc     = (const float*)d_in[10];
  float* out = (float*)d_out;
  char*  ws  = (char*)d_ws;

  // workspace layout (all offsets 256B aligned by construction)
  const size_t off_h    = 0;                                  // 2*64*300*4
  const size_t off_c    = off_h    + 2UL*BATCH*HID*4;         // 153600
  const size_t off_h16  = off_c    + 2UL*BATCH*HID*4;         // 307200
  const size_t off_hcat = off_h16  + 2UL*BATCH*KPAD*2;        // 389120
  const size_t zero_end = off_hcat + (size_t)ROWS*KCAT*2;     // 40235008
  const size_t off_x16  = zero_end;
  const size_t off_wpk  = off_x16  + (size_t)ROWS*KPAD*2;
  const size_t off_upk  = off_wpk  + 2UL*KT_K*NT_W*512*2;
  const size_t off_wfc  = off_upk  + 2UL*4*KT_K*NSTRIP*512*2;
  const size_t off_xg   = off_wfc  + (size_t)KT_C*4*512*2;

  float*    h_state = (float*)(ws + off_h);
  float*    c_state = (float*)(ws + off_c);
  _Float16* h16     = (_Float16*)(ws + off_h16);
  _Float16* hcat    = (_Float16*)(ws + off_hcat);
  _Float16* X16     = (_Float16*)(ws + off_x16);
  _Float16* Wpk     = (_Float16*)(ws + off_wpk);
  _Float16* Upk     = (_Float16*)(ws + off_upk);
  _Float16* Wfcpk   = (_Float16*)(ws + off_wfc);
  _Float16* xg      = (_Float16*)(ws + off_xg);

  // 1) zero state + hcat padding
  zero_ws<<<4096, 256, 0, stream>>>((f4*)ws, (long)(zero_end / 16));

  // 2) pack weights into WMMA B-fragment layout (f32 -> f16)
  pack_W  <<<3000, 256, 0, stream>>>(Wf, Wb, Wpk);
  pack_U  <<<3040, 256, 0, stream>>>(Uf, Ub, Upk);
  pack_Wfc<<< 152, 256, 0, stream>>>(Wfc, Wfcpk);

  // 3) embedding gather -> f16, K padded
  gather_x<<<8192, 256, 0, stream>>>(tokens, emb, X16);

  // 4) xg = X@W + b for both directions (A tile staged in LDS per block)
  gemm_xg<<<dim3((NT_W + 7) / 8, ROWS / 16, 2), 256, 0, stream>>>(
      X16, Wpk, bf, bb, xg);

  // 5) sequential scan, both directions per launch
  for (int s = 0; s < TLEN; ++s)
    lstm_step<<<dim3(NSTRIP, BATCH / 16, 2), 128, 0, stream>>>(
        xg, Upk, bf, bb, seq_len, h_state, c_state, h16, hcat, s);

  // 6) out = hcat @ Wfc + bfc
  gemm_fc<<<dim3(TAGS / 16, ROWS / 16 / 8, 1), 256, 0, stream>>>(hcat, Wfcpk, bfc, out);
}
